// SemanticsModulatedAttention_11278584120000
// MI455X (gfx1250) — matrix operand order repaired and checked by reading, not compiled, not run on hardware
//
#include <hip/hip_runtime.h>
#include <hip/hip_bf16.h>

#define BB 4
#define TT 4096
#define DD 1024
#define TDD 768
#define HH 16
#define KK 4
#define NTOK 4101        // 1 + K + T
#define NEGC (-1000000.0f)

typedef __attribute__((ext_vector_type(16))) __bf16 v16bf;
typedef __attribute__((ext_vector_type(8)))  float  v8f;

union Frag16 {
    v16bf v;
    unsigned short u[16];
    uint4 q[2];
};

__device__ __forceinline__ unsigned short f2bf(float f) {
    unsigned int u = __float_as_uint(f);
    unsigned int r = u + 0x7FFFu + ((u >> 16) & 1u);   // round-to-nearest-even
    return (unsigned short)(r >> 16);
}

// ---------------- f32 -> bf16 bulk convert ----------------
__global__ void cvt_bf16_kernel(const float* __restrict__ s, unsigned short* __restrict__ d, int n) {
    for (int i = blockIdx.x * blockDim.x + threadIdx.x; i < n; i += gridDim.x * blockDim.x)
        d[i] = f2bf(s[i]);
}

// ---------------- LayerNorm of x -> bf16 xn ----------------
__global__ __launch_bounds__(256) void ln_x_kernel(const float* __restrict__ x,
                                                   const float* __restrict__ g,
                                                   const float* __restrict__ bt,
                                                   unsigned short* __restrict__ xnbf) {
    __shared__ float red[256], red2[256];
    int row = blockIdx.x;
    int tid = threadIdx.x;
    const float* xr = x + (size_t)row * DD;
    float4 xv = *(const float4*)(xr + tid * 4);
    float s  = xv.x + xv.y + xv.z + xv.w;
    float s2 = xv.x * xv.x + xv.y * xv.y + xv.z * xv.z + xv.w * xv.w;
    red[tid] = s; red2[tid] = s2; __syncthreads();
    for (int o = 128; o > 0; o >>= 1) {
        if (tid < o) { red[tid] += red[tid + o]; red2[tid] += red2[tid + o]; }
        __syncthreads();
    }
    float mean = red[0] * (1.0f / DD);
    float var  = red2[0] * (1.0f / DD) - mean * mean;
    float rstd = rsqrtf(var + 1e-5f);
    int i0 = tid * 4;
    unsigned short h0 = f2bf((xv.x - mean) * rstd * g[i0 + 0] + bt[i0 + 0]);
    unsigned short h1 = f2bf((xv.y - mean) * rstd * g[i0 + 1] + bt[i0 + 1]);
    unsigned short h2 = f2bf((xv.z - mean) * rstd * g[i0 + 2] + bt[i0 + 2]);
    unsigned short h3 = f2bf((xv.w - mean) * rstd * g[i0 + 3] + bt[i0 + 3]);
    uint2 pk;
    pk.x = (unsigned int)h0 | ((unsigned int)h1 << 16);
    pk.y = (unsigned int)h2 | ((unsigned int)h3 << 16);
    *(uint2*)(xnbf + (size_t)row * DD + i0) = pk;
}

// ---------------- Main GEMM: C = xn @ W^T + bias, 32x32 tile per wave -----
// mode 0: q    -> q[r*D+c]            *= src
// mode 1: kmot -> key[b][5+t][c]      += (1-src)*NEG
// mode 2: vmot -> value[b][5+t][c]    *= src
__device__ __forceinline__ void store_tile(v8f acc, int row0, int col0, int half, int lcol,
                                           const float* __restrict__ bias,
                                           const int* __restrict__ srcmask,
                                           float* __restrict__ out, int mode) {
#pragma unroll
    for (int i = 0; i < 8; ++i) {
        int r = row0 + half * 8 + i;
        int c = col0 + lcol;
        float v = acc[i] + bias[c];
        int b_ = r >> 12;            // / T
        int t_ = r & 4095;
        float s = (float)srcmask[r];
        if (mode == 0) {
            out[(size_t)r * DD + c] = v * s;
        } else if (mode == 1) {
            out[((size_t)b_ * NTOK + 5 + t_) * DD + c] = v + (1.0f - s) * NEGC;
        } else {
            out[((size_t)b_ * NTOK + 5 + t_) * DD + c] = v * s;
        }
    }
}

__global__ __launch_bounds__(256) void gemm_qkv(const unsigned short* __restrict__ xnbf,
                                                const unsigned short* __restrict__ Wb,
                                                const float* __restrict__ bias,
                                                const int* __restrict__ srcmask,
                                                float* __restrict__ out, int mode) {
    int wave = threadIdx.x >> 5;
    int lane = threadIdx.x & 31;
    int tile = blockIdx.x * 8 + wave;       // 16384 tiles total (32x32 each)
    int mTile = tile >> 5;                  // 512 row tiles
    int nTile = tile & 31;                  // 32 col tiles
    int row0 = mTile * 32;
    int col0 = nTile * 32;
    int half = lane >> 4;
    int lcol = lane & 15;

    const unsigned short* aPtr0 = xnbf + (size_t)(row0 + lcol) * DD;
    const unsigned short* aPtr1 = aPtr0 + (size_t)16 * DD;
    const unsigned short* bPtr0 = Wb + (size_t)(col0 + lcol) * DD;
    const unsigned short* bPtr1 = bPtr0 + (size_t)16 * DD;

    v8f acc00 = {0.f,0.f,0.f,0.f,0.f,0.f,0.f,0.f};
    v8f acc01 = acc00, acc10 = acc00, acc11 = acc00;

    for (int k0 = 0; k0 < DD; k0 += 32) {
        __builtin_prefetch(aPtr0 + k0 + 256, 0, 3);
        __builtin_prefetch(aPtr1 + k0 + 256, 0, 3);
        Frag16 a0, a1, b0, b1;
        a0.q[0] = *(const uint4*)(aPtr0 + k0 + half * 8);
        a0.q[1] = *(const uint4*)(aPtr0 + k0 + 16 + half * 8);
        a1.q[0] = *(const uint4*)(aPtr1 + k0 + half * 8);
        a1.q[1] = *(const uint4*)(aPtr1 + k0 + 16 + half * 8);
        b0.q[0] = *(const uint4*)(bPtr0 + k0 + half * 8);
        b0.q[1] = *(const uint4*)(bPtr0 + k0 + 16 + half * 8);
        b1.q[0] = *(const uint4*)(bPtr1 + k0 + half * 8);
        b1.q[1] = *(const uint4*)(bPtr1 + k0 + 16 + half * 8);
        acc00 = __builtin_amdgcn_wmma_f32_16x16x32_bf16(false, a0.v, false, b0.v, (short)0, acc00, false, false);
        acc01 = __builtin_amdgcn_wmma_f32_16x16x32_bf16(false, a0.v, false, b1.v, (short)0, acc01, false, false);
        acc10 = __builtin_amdgcn_wmma_f32_16x16x32_bf16(false, a1.v, false, b0.v, (short)0, acc10, false, false);
        acc11 = __builtin_amdgcn_wmma_f32_16x16x32_bf16(false, a1.v, false, b1.v, (short)0, acc11, false, false);
    }
    store_tile(acc00, row0,      col0,      half, lcol, bias, srcmask, out, mode);
    store_tile(acc01, row0,      col0 + 16, half, lcol, bias, srcmask, out, mode);
    store_tile(acc10, row0 + 16, col0,      half, lcol, bias, srcmask, out, mode);
    store_tile(acc11, row0 + 16, col0 + 16, half, lcol, bias, srcmask, out, mode);
}

// ---------------- text streams: k_text / v_text (rows n=0) ----------------
__global__ __launch_bounds__(256) void text_streams(const float* __restrict__ xf,
                                                    const float* __restrict__ tg,
                                                    const float* __restrict__ tb,
                                                    const float* __restrict__ Wkt,
                                                    const float* __restrict__ bkt,
                                                    const float* __restrict__ Wvt,
                                                    const float* __restrict__ bvt,
                                                    const int* __restrict__ cond,
                                                    float* __restrict__ key,
                                                    float* __restrict__ value) {
    __shared__ float buf[TDD];
    __shared__ float red[256], red2[256];
    int b = blockIdx.x, tid = threadIdx.x;
    const float* xr = xf + (size_t)b * TDD;
    float s = 0.f, s2 = 0.f;
    for (int i = tid; i < TDD; i += 256) { float v = xr[i]; buf[i] = v; s += v; s2 += v * v; }
    red[tid] = s; red2[tid] = s2; __syncthreads();
    for (int o = 128; o > 0; o >>= 1) {
        if (tid < o) { red[tid] += red[tid + o]; red2[tid] += red2[tid + o]; }
        __syncthreads();
    }
    float mean = red[0] * (1.0f / TDD);
    float rstd = rsqrtf(red2[0] * (1.0f / TDD) - mean * mean + 1e-5f);
    for (int i = tid; i < TDD; i += 256) buf[i] = (buf[i] - mean) * rstd * tg[i] + tb[i];
    __syncthreads();
    float text_c = ((cond[b] % 10) > 0) ? 1.0f : 0.0f;
    for (int j = tid; j < DD; j += 256) {
        const float* wk = Wkt + (size_t)j * TDD;
        const float* wv = Wvt + (size_t)j * TDD;
        float ak = 0.f, av = 0.f;
        for (int i = 0; i < TDD; ++i) { float u = buf[i]; ak += u * wk[i]; av += u * wv[i]; }
        key  [((size_t)b * NTOK) * DD + j] = ak + bkt[j] + (1.0f - text_c) * NEGC;
        value[((size_t)b * NTOK) * DD + j] = (av + bvt[j]) * text_c;
    }
}

// ---------------- retrieval key stream (rows n=1..4) ----------------------
__global__ __launch_bounds__(256) void retr_key(const float* __restrict__ re_motion,
                                                const float* __restrict__ re_text,
                                                const float* __restrict__ r1g,
                                                const float* __restrict__ r1b,
                                                const float* __restrict__ Wkr,
                                                const float* __restrict__ bkr,
                                                const int* __restrict__ cond,
                                                float* __restrict__ key) {
    __shared__ float buf[2 * DD];
    __shared__ float red[256], red2[256];
    int b = blockIdx.x >> 2, k = blockIdx.x & 3, tid = threadIdx.x;
    const float* m = re_motion + (size_t)(b * KK + k) * DD;
    const float* t = re_text   + (size_t)(b * KK + k) * DD;
    float s = 0.f, s2 = 0.f;
    for (int i = tid; i < DD; i += 256) { float v = m[i]; buf[i] = v;      s += v; s2 += v * v; }
    for (int i = tid; i < DD; i += 256) { float v = t[i]; buf[DD + i] = v; s += v; s2 += v * v; }
    red[tid] = s; red2[tid] = s2; __syncthreads();
    for (int o = 128; o > 0; o >>= 1) {
        if (tid < o) { red[tid] += red[tid + o]; red2[tid] += red2[tid + o]; }
        __syncthreads();
    }
    float mean = red[0] * (1.0f / (2 * DD));
    float rstd = rsqrtf(red2[0] * (1.0f / (2 * DD)) - mean * mean + 1e-5f);
    for (int i = tid; i < 2 * DD; i += 256) buf[i] = (buf[i] - mean) * rstd * r1g[i] + r1b[i];
    __syncthreads();
    float retr_c = ((cond[b] / 10) > 0) ? 1.0f : 0.0f;
    for (int j = tid; j < DD; j += 256) {
        const float* w = Wkr + (size_t)j * 2 * DD;
        float a = 0.f;
        for (int i = 0; i < 2 * DD; ++i) a += buf[i] * w[i];
        key[((size_t)b * NTOK + 1 + k) * DD + j] = a + bkr[j] + (1.0f - retr_c) * NEGC;
    }
}

// ---------------- retrieval value stream (rows n=1..4) --------------------
__global__ __launch_bounds__(256) void retr_val(const float* __restrict__ re_motion,
                                                const float* __restrict__ r2g,
                                                const float* __restrict__ r2b,
                                                const float* __restrict__ Wvr,
                                                const float* __restrict__ bvr,
                                                const int* __restrict__ cond,
                                                float* __restrict__ value) {
    __shared__ float buf[DD];
    __shared__ float red[256], red2[256];
    int b = blockIdx.x >> 2, k = blockIdx.x & 3, tid = threadIdx.x;
    const float* m = re_motion + (size_t)(b * KK + k) * DD;
    float s = 0.f, s2 = 0.f;
    for (int i = tid; i < DD; i += 256) { float v = m[i]; buf[i] = v; s += v; s2 += v * v; }
    red[tid] = s; red2[tid] = s2; __syncthreads();
    for (int o = 128; o > 0; o >>= 1) {
        if (tid < o) { red[tid] += red[tid + o]; red2[tid] += red2[tid + o]; }
        __syncthreads();
    }
    float mean = red[0] * (1.0f / DD);
    float rstd = rsqrtf(red2[0] * (1.0f / DD) - mean * mean + 1e-5f);
    for (int i = tid; i < DD; i += 256) buf[i] = (buf[i] - mean) * rstd * r2g[i] + r2b[i];
    __syncthreads();
    float retr_c = ((cond[b] / 10) > 0) ? 1.0f : 0.0f;
    for (int j = tid; j < DD; j += 256) {
        const float* w = Wvr + (size_t)j * DD;
        float a = 0.f;
        for (int i = 0; i < DD; ++i) a += buf[i] * w[i];
        value[((size_t)b * NTOK + 1 + k) * DD + j] = (a + bvr[j]) * retr_c;
    }
}

// ---------------- softmax over head dim (q) -> bf16 -----------------------
__global__ __launch_bounds__(256) void softmax_q(const float* __restrict__ q,
                                                 unsigned short* __restrict__ qbf) {
    int wave = threadIdx.x >> 5, lane = threadIdx.x & 31;
    int g = blockIdx.x * 8 + wave;                 // group over B*T*H
    const float* p = q + (size_t)g * 64;
    float e0 = p[lane], e1 = p[lane + 32];
    float mx = fmaxf(e0, e1);
    for (int o = 16; o > 0; o >>= 1) mx = fmaxf(mx, __shfl_xor(mx, o, 32));
    e0 = __expf(e0 - mx); e1 = __expf(e1 - mx);
    float s = e0 + e1;
    for (int o = 16; o > 0; o >>= 1) s += __shfl_xor(s, o, 32);
    float inv = 1.0f / s;
    qbf[(size_t)g * 64 + lane]      = f2bf(e0 * inv);
    qbf[(size_t)g * 64 + lane + 32] = f2bf(e1 * inv);
}

// ---------------- softmax over tokens (k) -> bf16 -------------------------
__global__ __launch_bounds__(256) void softmax_k(const float* __restrict__ key,
                                                 unsigned short* __restrict__ kbf) {
    __shared__ float red[256];
    __shared__ float cm[64], cs[64];
    int b = blockIdx.x >> 4, h = blockIdx.x & 15;
    int col = threadIdx.x & 63, sub = threadIdx.x >> 6;
    const float* base = key + (size_t)b * NTOK * DD + h * 64 + col;
    float m = -3.0e38f;
    for (int n = sub; n < NTOK; n += 4) m = fmaxf(m, base[(size_t)n * DD]);
    red[threadIdx.x] = m; __syncthreads();
    if (sub == 0)
        cm[col] = fmaxf(fmaxf(red[col], red[col + 64]), fmaxf(red[col + 128], red[col + 192]));
    __syncthreads();
    float cmax = cm[col];
    float s = 0.f;
    for (int n = sub; n < NTOK; n += 4) s += __expf(base[(size_t)n * DD] - cmax);
    red[threadIdx.x] = s; __syncthreads();
    if (sub == 0)
        cs[col] = red[col] + red[col + 64] + red[col + 128] + red[col + 192];
    __syncthreads();
    float inv = 1.0f / cs[col];
    unsigned short* ob = kbf + (size_t)b * NTOK * DD + h * 64 + col;
    for (int n = sub; n < NTOK; n += 4)
        ob[(size_t)n * DD] = f2bf(__expf(base[(size_t)n * DD] - cmax) * inv);
}

// ---------------- attn = k^T @ v per (b,h): 64x64, K-dim NTOK -------------
__global__ __launch_bounds__(256) void attn_kernel(const unsigned short* __restrict__ kbf,
                                                   const unsigned short* __restrict__ vbf,
                                                   unsigned short* __restrict__ attnT) {
    int wave = threadIdx.x >> 5, lane = threadIdx.x & 31;
    int tile = blockIdx.x * 8 + wave;       // 1024 tiles
    int nt = tile & 3;
    int mt = (tile >> 2) & 3;
    int bh = tile >> 4;                     // 0..63
    int b = bh >> 4, h = bh & 15;
    int half = lane >> 4, lcol = lane & 15;
    int dA = h * 64 + mt * 16 + lcol;
    int dB = h * 64 + nt * 16 + lcol;
    const unsigned short* kb = kbf + (size_t)b * NTOK * DD;
    const unsigned short* vb = vbf + (size_t)b * NTOK * DD;

    v8f acc = {0.f,0.f,0.f,0.f,0.f,0.f,0.f,0.f};
    for (int n0 = 0; n0 < NTOK; n0 += 32) {
        Frag16 a, bf;
#pragma unroll
        for (int j = 0; j < 8; ++j) {
            int t0 = n0 + half * 8 + j;
            int t1 = n0 + 16 + half * 8 + j;
            a.u[j]     = (t0 < NTOK) ? kb[(size_t)t0 * DD + dA] : (unsigned short)0;
            a.u[j + 8] = (t1 < NTOK) ? kb[(size_t)t1 * DD + dA] : (unsigned short)0;
        }
#pragma unroll
        for (int j = 0; j < 16; ++j) {
            int t = n0 + half * 8 + (j & 7) + (j >> 3) * 16;
            bf.u[j] = (t < NTOK) ? vb[(size_t)t * DD + dB] : (unsigned short)0;
        }
        acc = __builtin_amdgcn_wmma_f32_16x16x32_bf16(false, a.v, false, bf.v,
                                                      (short)0, acc, false, false);
    }
#pragma unroll
    for (int i = 0; i < 8; ++i) {
        int r = mt * 16 + half * 8 + i;     // d index within 64
        int c = nt * 16 + lcol;             // l index within 64
        attnT[(size_t)bh * 4096 + (size_t)c * 64 + r] = f2bf(acc[i]);  // column-major attn
    }
}

// ---------------- y = q @ attn per (b,h): T x 64, K=64 --------------------
__global__ __launch_bounds__(256) void y_kernel(const unsigned short* __restrict__ qbf,
                                                const unsigned short* __restrict__ attnT,
                                                float* __restrict__ out) {
    int wave = threadIdx.x >> 5, lane = threadIdx.x & 31;
    int tile = blockIdx.x * 8 + wave;       // 65536 tiles
    int nt = tile & 3;
    int mt = (tile >> 2) & 255;
    int bh = tile >> 10;                    // 0..63
    int b = bh >> 4, h = bh & 15;
    int half = lane >> 4, lcol = lane & 15;
    int row = mt * 16 + lcol;
    const unsigned short* qrow = qbf + (size_t)(b * TT + row) * DD + h * 64;
    const unsigned short* at   = attnT + (size_t)bh * 4096 + (size_t)(nt * 16 + lcol) * 64;

    v8f acc = {0.f,0.f,0.f,0.f,0.f,0.f,0.f,0.f};
#pragma unroll
    for (int k0 = 0; k0 < 64; k0 += 32) {
        Frag16 a, bb;
        a.q[0]  = *(const uint4*)(qrow + k0 + half * 8);
        a.q[1]  = *(const uint4*)(qrow + k0 + 16 + half * 8);
        bb.q[0] = *(const uint4*)(at + k0 + half * 8);
        bb.q[1] = *(const uint4*)(at + k0 + 16 + half * 8);
        acc = __builtin_amdgcn_wmma_f32_16x16x32_bf16(false, a.v, false, bb.v,
                                                      (short)0, acc, false, false);
    }
#pragma unroll
    for (int i = 0; i < 8; ++i) {
        int r = mt * 16 + half * 8 + i;
        int c = h * 64 + nt * 16 + lcol;
        out[(size_t)(b * TT + r) * DD + c] = acc[i];
    }
}

extern "C" void kernel_launch(void* const* d_in, const int* in_sizes, int n_in,
                              void* d_out, int out_size, void* d_ws, size_t ws_size,
                              hipStream_t stream) {
    (void)in_sizes; (void)n_in; (void)out_size; (void)ws_size;
    const float* x         = (const float*)d_in[0];
    const float* xf        = (const float*)d_in[1];
    const int*   src_mask  = (const int*)d_in[2];
    const int*   cond_type = (const int*)d_in[3];
    const float* re_motion = (const float*)d_in[4];
    const float* re_text   = (const float*)d_in[5];
    // d_in[6] = num_heads (hardcoded HH=16)
    const float* ng  = (const float*)d_in[7];
    const float* nb  = (const float*)d_in[8];
    const float* tg  = (const float*)d_in[9];
    const float* tb  = (const float*)d_in[10];
    const float* r1g = (const float*)d_in[11];
    const float* r1b = (const float*)d_in[12];
    const float* r2g = (const float*)d_in[13];
    const float* r2b = (const float*)d_in[14];
    const float* Wq  = (const float*)d_in[15];
    const float* bq  = (const float*)d_in[16];
    const float* Wkt = (const float*)d_in[17];
    const float* bkt = (const float*)d_in[18];
    const float* Wvt = (const float*)d_in[19];
    const float* bvt = (const float*)d_in[20];
    const float* Wkm = (const float*)d_in[21];
    const float* bkm = (const float*)d_in[22];
    const float* Wvm = (const float*)d_in[23];
    const float* bvm = (const float*)d_in[24];
    const float* Wkr = (const float*)d_in[25];
    const float* bkr = (const float*)d_in[26];
    const float* Wvr = (const float*)d_in[27];
    const float* bvr = (const float*)d_in[28];
    float* out = (float*)d_out;

    char* ws = (char*)d_ws;
    const size_t szXN  = (size_t)BB * TT * DD * 2;            // 32 MB  bf16 xn
    const size_t szW   = (size_t)DD * DD * 2;                 //  2 MB  per weight
    const size_t szQ   = (size_t)BB * TT * DD * 4;            // 64 MB  f32 q
    const size_t szKV  = (size_t)BB * NTOK * DD * 4;          // 67 MB  f32 key/value
    const size_t szQB  = (size_t)BB * TT * DD * 2;            // 32 MB  bf16 q
    const size_t szKVB = (size_t)BB * NTOK * DD * 2;          // 33 MB  bf16 k/v

    unsigned short* xnbf  = (unsigned short*)(ws);
    unsigned short* wqbf  = (unsigned short*)(ws + szXN);
    unsigned short* wkmbf = (unsigned short*)(ws + szXN + szW);
    unsigned short* wvmbf = (unsigned short*)(ws + szXN + 2 * szW);
    float*          qf    = (float*)(ws + szXN + 3 * szW);
    float*          keyf  = (float*)(ws + szXN + 3 * szW + szQ);
    float*          valf  = (float*)(ws + szXN + 3 * szW + szQ + szKV);
    unsigned short* qbf   = (unsigned short*)(ws + szXN + 3 * szW + szQ + 2 * szKV);
    unsigned short* kbf   = (unsigned short*)(ws + szXN + 3 * szW + szQ + 2 * szKV + szQB);
    unsigned short* vbf   = (unsigned short*)(ws + szXN + 3 * szW + szQ + 2 * szKV + szQB + szKVB);
    unsigned short* attnT = (unsigned short*)(ws + szXN + 3 * szW + szQ + 2 * szKV + szQB + 2 * szKVB);

    // 1. weights -> bf16
    cvt_bf16_kernel<<<4096, 256, 0, stream>>>(Wq,  wqbf,  DD * DD);
    cvt_bf16_kernel<<<4096, 256, 0, stream>>>(Wkm, wkmbf, DD * DD);
    cvt_bf16_kernel<<<4096, 256, 0, stream>>>(Wvm, wvmbf, DD * DD);

    // 2. LayerNorm x -> xn bf16
    ln_x_kernel<<<BB * TT, 256, 0, stream>>>(x, ng, nb, xnbf);

    // 3. three big WMMA GEMMs (32x32 per wave, 4 wmma per K-step): q, k_mot, v_mot
    gemm_qkv<<<2048, 256, 0, stream>>>(xnbf, wqbf,  bq,  src_mask, qf,   0);
    gemm_qkv<<<2048, 256, 0, stream>>>(xnbf, wkmbf, bkm, src_mask, keyf, 1);
    gemm_qkv<<<2048, 256, 0, stream>>>(xnbf, wvmbf, bvm, src_mask, valf, 2);

    // 4. tiny streams -> key/value rows 0..4
    text_streams<<<BB, 256, 0, stream>>>(xf, tg, tb, Wkt, bkt, Wvt, bvt, cond_type, keyf, valf);
    retr_key<<<BB * KK, 256, 0, stream>>>(re_motion, re_text, r1g, r1b, Wkr, bkr, cond_type, keyf);
    retr_val<<<BB * KK, 256, 0, stream>>>(re_motion, r2g, r2b, Wvr, bvr, cond_type, valf);

    // 5. softmaxes
    softmax_q<<<(BB * TT * HH) / 8, 256, 0, stream>>>(qf, qbf);
    softmax_k<<<BB * HH, 256, 0, stream>>>(keyf, kbf);

    // 6. value -> bf16
    cvt_bf16_kernel<<<16384, 256, 0, stream>>>(valf, vbf, BB * NTOK * DD);

    // 7. attn = k^T v  (WMMA), stored transposed bf16
    attn_kernel<<<128, 256, 0, stream>>>(kbf, vbf, attnT);

    // 8. y = q attn (WMMA) -> f32 output
    y_kernel<<<8192, 256, 0, stream>>>(qbf, attnT, out);
}
